// RDCRNN_18030272709083
// MI455X (gfx1250) — matrix operand order repaired and checked by reading, not compiled, not run on hardware
//
#include <hip/hip_runtime.h>

#define Hdim 64
#define Wdim 64
#define HW   4096
#define Bsz  4
#define Tlen 8

#define WSLAB (9 * 5 * 32 * 16)   // bf16 halfs per weight chunk slab (23040)

typedef __attribute__((ext_vector_type(16))) __bf16 v16bf;
typedef __attribute__((ext_vector_type(8)))  float  v8f;
typedef __attribute__((ext_vector_type(4)))  unsigned int v4u;
typedef __attribute__((ext_vector_type(8)))  int v8i_t;
typedef __attribute__((ext_vector_type(4)))  int v4i_t;

__device__ __forceinline__ unsigned short f2bf(float f) {
  unsigned u = __float_as_uint(f);
  u += 0x7fffu + ((u >> 16) & 1u);         // round-to-nearest-even
  return (unsigned short)(u >> 16);
}

// ---------------------------------------------------------------------------
// Pack conv weights (65, ci, 3, 3) fp32 -> bf16, pre-swizzled into the per-lane
// A-fragment layout of v_wmma_f32_16x16x32_bf16 (M=out-channel, K=in-channel
// within a 32-channel chunk). dst layout: [q][p=ky*3+kx][mt][lane][h] (16 bf16
// per lane = one contiguous 32B fragment read). Output channels padded 65->80.
// ---------------------------------------------------------------------------
__global__ void pack_w_kernel(const float* __restrict__ w,
                              unsigned short* __restrict__ dst,
                              int ci, int total) {
  int idx = blockIdx.x * blockDim.x + threadIdx.x;
  if (idx >= total) return;
  int h    = idx & 15;
  int lane = (idx >> 4) & 31;
  int mt   = (idx >> 9) % 5;
  int p    = (idx / (16 * 32 * 5)) % 9;
  int q    = idx / (16 * 32 * 5 * 9);
  // contiguous K order used here (same as B): lanes0-15 K=0..15, lanes16-31 K=16..31
  int k  = h + ((lane >= 16) ? 16 : 0);
  int co = 16 * mt + (lane & 15);
  int cin_ch = 32 * q + k;
  int ky = p / 3, kx = p % 3;
  float v = 0.f;
  if (co < 65) v = w[((long)(co * ci + cin_ch) * 3 + ky) * 3 + kx];
  dst[idx] = f2bf(v);
}

// TDM weight-slab load: 1-D tensor, 46080 bytes = 5760 x 8B elements.
__device__ __forceinline__ void tdm_load_weights(const unsigned short* gsrc,
                                                 unsigned lds_byte_addr) {
  unsigned long long ga = (unsigned long long)(uintptr_t)gsrc;
  v4u g0;
  g0.x = 1u;                                     // count=1 (valid descriptor)
  g0.y = lds_byte_addr;                          // lds_addr [63:32]
  g0.z = (unsigned)(ga & 0xffffffffu);           // global_addr lo
  g0.w = (unsigned)((ga >> 32) & 0x01ffffffu) | 0x80000000u; // ga hi | type=2
  v8i_t g1;
  g1[0] = 0x00030000;            // wg_mask=0, data_size=3 (8B)
  g1[1] = (int)(5760u << 16);    // tensor_dim0 lo16 in [31:16]
  g1[2] = 0x00010000;            // tensor_dim0 hi=0 | tensor_dim1=1
  g1[3] = (int)(5760u << 16);    // tensor_dim1 hi=0 | tile_dim0=5760
  g1[4] = 0x00000001;            // tile_dim1=1, tile_dim2=0
  g1[5] = 5760;                  // tensor_dim0_stride lo32
  g1[6] = 0;                     // stride0 hi, stride1 lo
  g1[7] = 0;
  v4i_t gz = {0, 0, 0, 0};
#if defined(__clang_major__) && (__clang_major__ >= 23)
  v8i_t gz8 = {0, 0, 0, 0, 0, 0, 0, 0};
  __builtin_amdgcn_tensor_load_to_lds(g0, g1, gz, gz, gz8, 0);
#else
  __builtin_amdgcn_tensor_load_to_lds(g0, g1, gz, gz, 0);
#endif
}

// ---------------------------------------------------------------------------
// Fused conv(3x3, ci=cin+32) + tanh + gate + cell update for one timestep.
// Block = one (b, y) image row (64 px); 4 waves, each: 80 channels x 16 px.
// GEMM: M=channels(80), N=pixels(16/wave), K=(cin+32)*9 in 32-ch chunks x 9 taps.
// Weight slabs are TDM-pipelined one chunk ahead into double-buffered LDS.
// ---------------------------------------------------------------------------
__global__ __launch_bounds__(128)
void conv_cell_kernel(const float* __restrict__ xin, int cin, long in_b_stride,
                      const float* __restrict__ cprev,        // null => zeros
                      const unsigned short* __restrict__ wpk, // packed weights
                      const float* __restrict__ bias,         // 65 fp32
                      float* __restrict__ outh, long outh_b_stride,
                      float* __restrict__ cnext,
                      float* __restrict__ lrh,   // null unless final fwd step
                      float* __restrict__ lrc,
                      float* __restrict__ lrg) {
  __shared__ __align__(32) unsigned short lds_in[3 * 66 * 32];      // 12.4 KB
  __shared__ __align__(32) unsigned short lds_w[2][WSLAB];          // 2x45 KB

  const int tid  = threadIdx.x;
  const int lane = tid & 31;
  const int wv   = tid >> 5;
  const int b    = blockIdx.x >> 6;
  const int y    = blockIdx.x & 63;
  const int x0w  = wv * 16;
  const int nchunks = (cin >> 5) + 1;   // last chunk = recurrent c state

  v8f acc[5];
#pragma unroll
  for (int i = 0; i < 5; ++i) acc[i] = v8f{};

  // prologue: kick off TDM for chunk 0
  if (wv == 0)
    tdm_load_weights(wpk, (unsigned)(uintptr_t)(void*)&lds_w[0][0]);

  for (int q = 0; q < nchunks; ++q) {
    const bool is_c = (q == nchunks - 1);
    // ---- stage input halo slab, transposed to [row][col][c] bf16 ----
    if (tid < 96) {
      int row = tid >> 5;            // 0..2  (y-1, y, y+1)
      int cc  = tid & 31;            // channel within chunk
      int yy  = y + row - 1;
      const float* src = nullptr;
      if (!is_c)      src = xin + (long)b * in_b_stride + (long)(32 * q + cc) * HW;
      else if (cprev) src = cprev + ((long)b * 32 + cc) * HW;
      const bool rowok = (yy >= 0) && (yy < Hdim) && (src != nullptr);
      lds_in[(row * 66 + 0)  * 32 + cc] = 0;    // left halo (x=-1)
      lds_in[(row * 66 + 65) * 32 + cc] = 0;    // right halo (x=64)
      if (rowok) {
        const float4* s4 = (const float4*)(src + (long)yy * Wdim);
        if (!is_c && q + 1 < nchunks - 1)
          __builtin_prefetch(src + HW * 32, 0, 1);   // next chunk, same ch (plane start, row 0)
#pragma unroll 4
        for (int c4 = 0; c4 < 16; ++c4) {
          float4 v = s4[c4];
          int base = (row * 66 + 1 + c4 * 4) * 32 + cc;
          lds_in[base]      = f2bf(v.x);
          lds_in[base + 32] = f2bf(v.y);
          lds_in[base + 64] = f2bf(v.z);
          lds_in[base + 96] = f2bf(v.w);
        }
      } else {
        for (int col = 1; col < 65; ++col) lds_in[(row * 66 + col) * 32 + cc] = 0;
      }
    }
    // ---- TDM pipeline: prefetch next slab, wait for current (in-order cnt) --
    if (wv == 0) {
      if (q + 1 < nchunks) {
        tdm_load_weights(wpk + (long)(q + 1) * WSLAB,
                         (unsigned)(uintptr_t)(void*)&lds_w[(q + 1) & 1][0]);
        __builtin_amdgcn_s_wait_tensorcnt(1);   // chunk q landed; q+1 in flight
      } else {
        __builtin_amdgcn_s_wait_tensorcnt(0);
      }
    }
    __syncthreads();

    const unsigned short* wbuf = &lds_w[q & 1][0];
    const int cbase = (lane >= 16) ? 16 : 0;   // B-matrix K-half per lane group
#pragma unroll
    for (int ky = 0; ky < 3; ++ky) {
#pragma unroll
      for (int kx = 0; kx < 3; ++kx) {
        const int p = ky * 3 + kx;
        const int colidx = x0w + (lane & 15) + kx;     // halo col (x-1 offset)
        const v16bf bfrag = *(const v16bf*)&lds_in[(ky * 66 + colidx) * 32 + cbase];
#pragma unroll
        for (int mt = 0; mt < 5; ++mt) {
          const v16bf afrag = *(const v16bf*)&wbuf[((p * 5 + mt) * 32 + lane) * 16];
          acc[mt] = __builtin_amdgcn_wmma_f32_16x16x32_bf16(
              false, afrag, false, bfrag, (short)0, acc[mt], false, false);
        }
      }
    }
    __syncthreads();
  }

  // ---- epilogue: bias + tanh, gate, cell update, stores ----
  const int mrow_off = (lane >= 16) ? 8 : 0;
  const int px = lane & 15;
  const int x  = x0w + px;
  float act[5][8];
#pragma unroll
  for (int mt = 0; mt < 5; ++mt)
#pragma unroll
    for (int r = 0; r < 8; ++r) {
      int ch = 16 * mt + r + mrow_off;
      float v = acc[mt][r];
      act[mt][r] = (ch < 65) ? tanhf(v + bias[ch]) : 0.f;
    }
  // gate = sigmoid(A[ch=64]) lives in mt=4, r=0, lanes 0..15 (one per pixel)
  float g  = 1.f / (1.f + __expf(-act[4][0]));
  float gv = __shfl(g, px, 32);                 // broadcast gate to my pixel

  const long spat = (long)y * Wdim + x;
#pragma unroll
  for (int mt = 0; mt < 2; ++mt)                // h channels 0..31
#pragma unroll
    for (int r = 0; r < 8; ++r) {
      int ch = 16 * mt + r + mrow_off;
      float hv = act[mt][r];
      outh[(long)b * outh_b_stride + (long)ch * HW + spat] = hv;
      if (lrh) lrh[((long)b * 32 + ch) * HW + spat] = hv;
    }
#pragma unroll
  for (int mt = 2; mt < 4; ++mt)                // cand 32..63 -> c_next
#pragma unroll
    for (int r = 0; r < 8; ++r) {
      int cch = 16 * (mt - 2) + r + mrow_off;
      float cn = gv * act[mt][r];
      cnext[((long)b * 32 + cch) * HW + spat] = cn;
      if (lrc) lrc[((long)b * 32 + cch) * HW + spat] = cn;
    }
  if (lrg && lane < 16)
    lrg[(long)b * HW + (long)y * Wdim + x0w + lane] = g;
}

// ---------------------------------------------------------------------------
extern "C" void kernel_launch(void* const* d_in, const int* in_sizes, int n_in,
                              void* d_out, int out_size, void* d_ws, size_t ws_size,
                              hipStream_t stream) {
  (void)in_sizes; (void)n_in; (void)out_size; (void)ws_size;
  const float* x = (const float*)d_in[0];
  float* out = (float*)d_out;                         // (T,B,256,H,W)
  float* lrh = out + (long)Tlen * Bsz * 256 * HW;     // (4,B,32,H,W)
  float* lrc = lrh + 4L * Bsz * 32 * HW;
  float* lrg = lrc + 4L * Bsz * 32 * HW;              // (4,B,1,H,W)

  const int chunks[4] = {2, 3, 5, 7};                 // (cin+32)/32 per layer
  const int ci_arr[4] = {64, 96, 160, 224};

  unsigned short* wpack = (unsigned short*)d_ws;
  long woff[8]; long tot = 0;
  for (int l = 0; l < 4; ++l)
    for (int d = 0; d < 2; ++d) { woff[l * 2 + d] = tot; tot += (long)chunks[l] * WSLAB; }
  float* cA = (float*)((char*)d_ws + tot * sizeof(unsigned short));
  float* cB = cA + (long)Bsz * 32 * HW;               // double-buffered c state

  // weight repack (deterministic; re-run every call)
  for (int l = 0; l < 4; ++l)
    for (int d = 0; d < 2; ++d) {
      const float* w = (const float*)d_in[1 + 4 * l + 2 * d];
      int total = chunks[l] * WSLAB;
      pack_w_kernel<<<(total + 255) / 256, 256, 0, stream>>>(
          w, wpack + woff[l * 2 + d], ci_arr[l], total);
    }

  for (int l = 0; l < 4; ++l) {
    const float* in_base; int cin; long ibs, its;
    if (l == 0) { in_base = x;   cin = 32;     ibs = 32L * HW;  its = (long)Bsz * 32 * HW; }
    else        { in_base = out; cin = 64 * l; ibs = 256L * HW; its = (long)Bsz * 256 * HW; }
    for (int d = 0; d < 2; ++d) {
      const unsigned short* wpk = wpack + woff[l * 2 + d];
      const float* bias = (const float*)d_in[2 + 4 * l + 2 * d];
      float* c_prev = nullptr; float* c_next = cA;
      for (int s = 0; s < Tlen; ++s) {
        int t = (d == 0) ? s : (Tlen - 1 - s);
        const float* xin_t = in_base + (long)t * its;
        float* outh = out + (long)t * Bsz * 256 * HW + (long)(64 * l + 32 * d) * HW;
        bool fin = (d == 0 && s == Tlen - 1);
        conv_cell_kernel<<<Bsz * Hdim, 128, 0, stream>>>(
            xin_t, cin, ibs, c_prev, wpk, bias, outh, 256L * HW, c_next,
            fin ? lrh + (long)l * Bsz * 32 * HW : nullptr,
            fin ? lrc + (long)l * Bsz * 32 * HW : nullptr,
            fin ? lrg + (long)l * Bsz * HW : nullptr);
        c_prev = c_next;
        c_next = (c_prev == cA) ? cB : cA;
      }
    }
  }
}
